// TNO_83794811945290
// MI455X (gfx1250) — matrix operand order repaired and checked by reading, not compiled, not run on hardware
//
#include <hip/hip_runtime.h>

// ---------------------------------------------------------------------------
// Problem constants (from reference)
// ---------------------------------------------------------------------------
static constexpr int LSEQ = 2048;
static constexpr int BB   = 4;
static constexpr int EE   = 1024;
static constexpr int HH   = 8;
static constexpr int D1   = 2 * EE;        // 2048
static constexpr int HD   = D1 / HH;       // 256
static constexpr int MROWS = LSEQ * BB;    // 8192
static constexpr int TLEN  = 4096;         // padded (reversed) toeplitz line/head

typedef __attribute__((ext_vector_type(16))) __bf16 v16bf;
typedef __attribute__((ext_vector_type(8)))  __bf16 v8bf;
typedef __attribute__((ext_vector_type(8)))  float  v8f;

__device__ __forceinline__ float silu_f(float x) {
    return x / (1.0f + __expf(-x));
}

__device__ __forceinline__ v16bf combine8(v8bf lo, v8bf hi) {
    return __builtin_shufflevector(lo, hi,
            0,1,2,3,4,5,6,7,8,9,10,11,12,13,14,15);
}

// ---------------------------------------------------------------------------
// f32 (N x K) -> bf16 transposed (K x N), LDS-tiled for coalescing both sides
// ---------------------------------------------------------------------------
__global__ void cvt_transpose_kernel(const float* __restrict__ src, // N x K
                                     __bf16* __restrict__ dst,      // K x N
                                     int N, int K) {
    __shared__ float tile[32][33];
    const int n0 = blockIdx.x * 32, k0 = blockIdx.y * 32;
    const int tx = threadIdx.x, ty = threadIdx.y;    // 32 x 8
    for (int r = ty; r < 32; r += 8)
        tile[r][tx] = src[(size_t)(n0 + r) * K + k0 + tx];
    __syncthreads();
    for (int r = ty; r < 32; r += 8)
        dst[(size_t)(k0 + r) * N + n0 + tx] = (__bf16)tile[tx][r];
}

// ---------------------------------------------------------------------------
// Build REVERSED per-head Toeplitz coefficient line:
//   t_rev[h][x] = t_{k-(L-1)} with k = TLEN-1-x
// so that A[m][k] = t_{(i-j)} = t_rev[2048 - i0 - m' + j0 + k] is ascending in k.
// ---------------------------------------------------------------------------
__global__ void build_t_kernel(const float* __restrict__ pos,
                               const float* __restrict__ zero,
                               const float* __restrict__ neg,
                               __bf16* __restrict__ t_rev) {
    int idx = blockIdx.x * 256 + threadIdx.x;
    if (idx >= HH * TLEN) return;
    int h = idx >> 12;
    int x = idx & (TLEN - 1);
    int k = (TLEN - 1) - x;
    int off = k - (LSEQ - 1);
    float val = 0.0f;
    if (off == 0)                         val = zero[h];
    else if (off > 0 && off <= LSEQ - 1)  val = pos[h * (LSEQ - 1) + off - 1];
    else if (off < 0)                     val = neg[h * (LSEQ - 1) + (-off) - 1];
    t_rev[idx] = (__bf16)val;
}

// ---------------------------------------------------------------------------
// SimpleRMSNorm: one block (256 thr) per row of 1024; output bf16
// ---------------------------------------------------------------------------
__global__ void rmsnorm_kernel(const float* __restrict__ q,
                               __bf16* __restrict__ x) {
    __shared__ float red[256];
    const int row = blockIdx.x;
    const float* qr = q + (size_t)row * EE;
    float s = 0.0f;
    for (int e = threadIdx.x; e < EE; e += 256) {
        float v = qr[e];
        s += v * v;
    }
    red[threadIdx.x] = s;
    __syncthreads();
    for (int off = 128; off > 0; off >>= 1) {
        if (threadIdx.x < off) red[threadIdx.x] += red[threadIdx.x + off];
        __syncthreads();
    }
    const float rms   = sqrtf(red[0]) * 0.03125f;   // * 1/sqrt(1024)
    const float scale = 1.0f / (rms + 1e-8f);
    __bf16* xr = x + (size_t)row * EE;
    for (int e = threadIdx.x; e < EE; e += 256) xr[e] = (__bf16)(qr[e] * scale);
}

// ---------------------------------------------------------------------------
// bf16 WMMA GEMM:  out[m][n] = epilogue( sum_k A[m][k] * Wt[k][n] )
//   A:  MROWS x KDIM row-major bf16;  Wt: KDIM x NDIM row-major bf16.
//   All dims compile-time -> strided epilogue stores use immediate offsets.
//   Wave tile: 32(m) x 64(n), 8 accumulators, A operands vector loads, B gathered per element.
//   MODE 0: silu(+bias) -> bf16 row-major            (u projection)
//   MODE 1: silu(+bias) -> bf16 permuted (b,h,n,d)   (v projection)
//   MODE 2: +bias +residual -> f32 row-major         (output projection)
// ---------------------------------------------------------------------------
template <int MODE, int NDIM, int KDIM>
__global__ void gemm_kernel(const __bf16* __restrict__ A,
                            const __bf16* __restrict__ Wt,
                            const float*  __restrict__ bias,
                            const float*  __restrict__ resid,
                            void* __restrict__ out) {
    constexpr int TILES_N = NDIM / 64;
    const int wave = threadIdx.x >> 5;
    const int lane = threadIdx.x & 31;
    const int gw   = blockIdx.x * (blockDim.x >> 5) + wave;
    const int m0 = (gw / TILES_N) * 32;
    const int n0 = (gw % TILES_N) * 64;

    v8f acc[2][4] = {};
    const int khalf8 = (lane >> 4) << 3;       // +8 for upper half-wave
    const __bf16* arow0 = A + (size_t)(m0 + (lane & 15)) * KDIM;
    const __bf16* arow1 = arow0 + (size_t)16 * KDIM;
    const __bf16* wbase = Wt + n0 + (lane & 15);   // lane supplies N

#pragma unroll 2
    for (int k0 = 0; k0 < KDIM; k0 += 32) {
        v16bf af0 = combine8(*(const v8bf*)(arow0 + k0 + khalf8),
                             *(const v8bf*)(arow0 + k0 + 16 + khalf8));
        v16bf af1 = combine8(*(const v8bf*)(arow1 + k0 + khalf8),
                             *(const v8bf*)(arow1 + k0 + 16 + khalf8));
        const __bf16* wr = wbase + (size_t)k0 * NDIM;
#pragma unroll
        for (int f = 0; f < 4; ++f) {
            v16bf bfv;
#pragma unroll
            for (int s = 0; s < 16; ++s) {
                const int kk = (s < 8) ? (khalf8 + s) : (16 + khalf8 + s - 8);
                bfv[s] = wr[(size_t)kk * NDIM + f * 16];
            }
            acc[0][f] = __builtin_amdgcn_wmma_f32_16x16x32_bf16(
                false, af0, false, bfv, (short)0, acc[0][f], false, false);
            acc[1][f] = __builtin_amdgcn_wmma_f32_16x16x32_bf16(
                false, af1, false, bfv, (short)0, acc[1][f], false, false);
        }
    }

    // ---- Epilogue (C/D: VGPR r -> M = r + 8*(lane>=16), N = lane&15) ----
    const int cm = (lane >> 4) * 8;
    const int cn = lane & 15;
    float bvals[4];
#pragma unroll
    for (int f = 0; f < 4; ++f) bvals[f] = bias[n0 + f * 16 + cn];

#pragma unroll
    for (int mi = 0; mi < 2; ++mi) {
#pragma unroll
        for (int f = 0; f < 4; ++f) {
#pragma unroll
            for (int r = 0; r < 8; ++r) {
                const int m = m0 + mi * 16 + r + cm;
                const int n = n0 + f * 16 + cn;
                float val = acc[mi][f][r] + bvals[f];
                if (MODE == 0) {
                    ((__bf16*)out)[(size_t)m * NDIM + n] = (__bf16)silu_f(val);
                } else if (MODE == 1) {
                    const int nn = m / BB, b = m % BB;     // m = nn*B + b
                    const int h  = n / HD, d = n % HD;     // n = h*HD + d
                    ((__bf16*)out)[(((size_t)(b * HH + h) * LSEQ + nn) * HD) + d] =
                        (__bf16)silu_f(val);
                } else {
                    ((float*)out)[(size_t)m * NDIM + n] =
                        val + resid[(size_t)m * NDIM + n];
                }
            }
        }
    }
}

// ---------------------------------------------------------------------------
// Toeplitz mix as dense WMMA matmul, fused with uy = u * y.
//   y[bh][i][d] = sum_j t[(i-j)] * v[bh][j][d]
//   Reversed t line (8KB bf16) staged in LDS -> A-fragments are plain
//   ascending contiguous runs (no lane packing); B-fragments are per-element
//   gathers down v columns.  Wave tile: 32(i) x 64(d), K-loop over 2048.
// ---------------------------------------------------------------------------
__global__ void toeplitz_kernel(const __bf16* __restrict__ t_rev,  // H x 4096
                                const __bf16* __restrict__ v_arr,  // (B*H) x L x HD
                                const __bf16* __restrict__ u_bf,   // M x D1
                                __bf16* __restrict__ uy) {         // M x D1
    __shared__ __bf16 t_s[TLEN];
    const int bh = blockIdx.x;           // bh = b*H + h
    const int h  = bh % HH;
    const int b  = bh / HH;
    const int tid = threadIdx.x;

    {   // cooperative 8KB stage: 512 uint4
        const uint4* src = (const uint4*)(t_rev + (size_t)h * TLEN);
        uint4* dst = (uint4*)t_s;
        for (int i = tid; i < 512; i += 256) dst[i] = src[i];
    }
    __syncthreads();

    const int wave = tid >> 5, lane = tid & 31;
    const int wtile  = blockIdx.y * 8 + wave;   // 0..255
    const int itile  = wtile >> 2;              // 0..63  (32-row tiles)
    const int dstrip = wtile & 3;               // 0..3   (64-col strips)
    const int i0 = itile * 32, d0 = dstrip * 64;

    v8f acc[2][4] = {};
    const int am     = lane & 15;
    const int khalf8 = (lane >> 4) << 3;
    const int c0     = LSEQ - i0 - am;          // A[m][k] = t_s[c0 + j0 - m16 + k]
    const __bf16* vb = v_arr + (size_t)bh * LSEQ * HD;

#pragma unroll 2
    for (int j0 = 0; j0 < LSEQ; j0 += 32) {
        // ---- A fragments: ascending contiguous runs from reversed line ----
        const __bf16* tr0 = t_s + (c0 + j0);        // rows i0..i0+15
        const __bf16* tr1 = tr0 - 16;               // rows i0+16..i0+31
        v16bf af0 = combine8(*(const v8bf*)(tr0 + khalf8),
                             *(const v8bf*)(tr0 + 16 + khalf8));
        v16bf af1 = combine8(*(const v8bf*)(tr1 + khalf8),
                             *(const v8bf*)(tr1 + 16 + khalf8));
        // ---- B fragments: lane supplies column d0+(lane&15), k gathered ----
        const __bf16* vrow = vb + (size_t)j0 * HD + d0 + (lane & 15);
#pragma unroll
        for (int f = 0; f < 4; ++f) {
            v16bf bfv;
#pragma unroll
            for (int s = 0; s < 16; ++s) {
                const int kk = (s < 8) ? (khalf8 + s) : (16 + khalf8 + s - 8);
                bfv[s] = vrow[(size_t)kk * HD + f * 16];
            }
            acc[0][f] = __builtin_amdgcn_wmma_f32_16x16x32_bf16(
                false, af0, false, bfv, (short)0, acc[0][f], false, false);
            acc[1][f] = __builtin_amdgcn_wmma_f32_16x16x32_bf16(
                false, af1, false, bfv, (short)0, acc[1][f], false, false);
        }
    }

    // ---- Epilogue: uy[m][fcol] = u[m][fcol] * y ----
    const int cm = (lane >> 4) * 8;
    const int cn = lane & 15;
#pragma unroll
    for (int mi = 0; mi < 2; ++mi) {
#pragma unroll
        for (int f = 0; f < 4; ++f) {
#pragma unroll
            for (int r = 0; r < 8; ++r) {
                const int i = i0 + mi * 16 + r + cm;
                const int d = d0 + f * 16 + cn;
                const size_t idx = (size_t)(i * BB + b) * D1 + (h * HD + d);
                const float uval = (float)u_bf[idx];
                uy[idx] = (__bf16)(uval * acc[mi][f][r]);
            }
        }
    }
}

// ---------------------------------------------------------------------------
// Launch
// ---------------------------------------------------------------------------
extern "C" void kernel_launch(void* const* d_in, const int* in_sizes, int n_in,
                              void* d_out, int out_size, void* d_ws, size_t ws_size,
                              hipStream_t stream) {
    (void)in_sizes; (void)n_in; (void)out_size; (void)ws_size;

    const float* q    = (const float*)d_in[0];
    // d_in[1] (key) and d_in[2] (value) are unused by the reference.
    const float* u_w  = (const float*)d_in[3];
    const float* u_b  = (const float*)d_in[4];
    const float* v_w  = (const float*)d_in[5];
    const float* v_b  = (const float*)d_in[6];
    const float* o_w  = (const float*)d_in[7];
    const float* o_b  = (const float*)d_in[8];
    const float* pos  = (const float*)d_in[9];
    const float* zero = (const float*)d_in[10];
    const float* neg  = (const float*)d_in[11];

    // ---- workspace carve-up ----
    char* ws = (char*)d_ws;
    size_t off = 0;
    auto carve = [&](size_t bytes) { char* p = ws + off; off += (bytes + 255) & ~(size_t)255; return p; };
    __bf16* x_bf  = (__bf16*)carve((size_t)MROWS * EE * 2);   // 16 MB
    __bf16* wtu   = (__bf16*)carve((size_t)EE * D1 * 2);      //  4 MB (E x D1)
    __bf16* wtv   = (__bf16*)carve((size_t)EE * D1 * 2);      //  4 MB (E x D1)
    __bf16* wto   = (__bf16*)carve((size_t)D1 * EE * 2);      //  4 MB (D1 x E)
    __bf16* t_rev = (__bf16*)carve((size_t)HH * TLEN * 2);    // 64 KB
    __bf16* u_bf  = (__bf16*)carve((size_t)MROWS * D1 * 2);   // 32 MB
    __bf16* v_arr = (__bf16*)carve((size_t)MROWS * D1 * 2);   // 32 MB
    __bf16* uy_bf = (__bf16*)carve((size_t)MROWS * D1 * 2);   // 32 MB

    // ---- prep: transposed bf16 weights + reversed toeplitz line ----
    {
        dim3 blk(32, 8);
        cvt_transpose_kernel<<<dim3(D1 / 32, EE / 32), blk, 0, stream>>>(u_w, wtu, D1, EE);
        cvt_transpose_kernel<<<dim3(D1 / 32, EE / 32), blk, 0, stream>>>(v_w, wtv, D1, EE);
        cvt_transpose_kernel<<<dim3(EE / 32, D1 / 32), blk, 0, stream>>>(o_w, wto, EE, D1);
        build_t_kernel<<<(HH * TLEN + 255) / 256, 256, 0, stream>>>(pos, zero, neg, t_rev);
    }

    // ---- RMSNorm ----
    rmsnorm_kernel<<<MROWS, 256, 0, stream>>>(q, x_bf);

    // ---- u = silu(x @ u_w^T + u_b);  v = silu(x @ v_w^T + v_b) (permuted) --
    {
        constexpr int blocks = (MROWS / 32) * (D1 / 64) / 8;   // 1024
        gemm_kernel<0, D1, EE><<<blocks, 256, 0, stream>>>(x_bf, wtu, u_b, nullptr, (void*)u_bf);
        gemm_kernel<1, D1, EE><<<blocks, 256, 0, stream>>>(x_bf, wtv, v_b, nullptr, (void*)v_arr);
    }

    // ---- y = T v (dense WMMA toeplitz), fused uy = u * y ----
    {
        dim3 grid(BB * HH, 32);                                // 32 x 32 blocks
        toeplitz_kernel<<<grid, 256, 0, stream>>>(t_rev, v_arr, u_bf, uy_bf);
    }

    // ---- out = uy @ o_w^T + o_b + query ----
    {
        constexpr int blocks = (MROWS / 32) * (EE / 64) / 8;   // 512
        gemm_kernel<2, EE, D1><<<blocks, 256, 0, stream>>>(uy_bf, wto, o_b, q, d_out);
    }
}
